// LSTMGenerator_14645838479708
// MI455X (gfx1250) — matrix operand order repaired and checked by reading, not compiled, not run on hardware
//
#include <hip/hip_runtime.h>
#include <hip/hip_bf16.h>
#include <stdint.h>

// ---------------------------------------------------------------------------
// LSTM generator for MI455X (gfx1250, wave32, WMMA + async-to-LDS).
//   B=256, S=512, IN=32, H0=64, H1=128, H2=256, OUT=1
// Phase 1 per layer: input GEMM (v_wmma_f32_16x16x32_f16), result stored in
//   fp16 *C-fragment order* so phase 2 loads accumulators with one b128.
// Phase 2 per layer: persistent recurrence, 16 workgroups (16-batch tiles),
//   H/32 waves each owning a 32-wide hidden slice; c in VGPRs, h in
//   double-buffered LDS, Whh streamed from L2 each step, next step's `pre`
//   fragments prefetched into LDS with GLOBAL_LOAD_ASYNC_TO_LDS_B128.
// ---------------------------------------------------------------------------

typedef __attribute__((ext_vector_type(16))) _Float16 v16h;
typedef __attribute__((ext_vector_type(8)))  _Float16 v8h;
typedef __attribute__((ext_vector_type(8)))  float    v8f;
typedef __attribute__((ext_vector_type(4)))  int      v4i;

#define BB 256
#define SS 512

// ---- gfx1250 async load-to-LDS (ASYNCcnt) --------------------------------
#if defined(__has_builtin)
#if __has_builtin(__builtin_amdgcn_global_load_async_to_lds_b128)
#define HAVE_ASYNC_LDS 1
#endif
#endif

static __device__ __forceinline__ void async_copy16(const _Float16* g, _Float16* l) {
#if defined(HAVE_ASYNC_LDS)
    __builtin_amdgcn_global_load_async_to_lds_b128(
        (__attribute__((address_space(1))) v4i*)g,
        (__attribute__((address_space(3))) v4i*)l,
        0, 0);
#else
    *(v8h*)l = *(const v8h*)g;
#endif
}

#if defined(HAVE_ASYNC_LDS)
#if __has_builtin(__builtin_amdgcn_s_wait_asynccnt)
#define ASYNC_WAIT(N) __builtin_amdgcn_s_wait_asynccnt(N)
#else
#define ASYNC_WAIT(N) asm volatile("s_wait_asynccnt %0" ::"i"(N) : "memory")
#endif
#else
#define ASYNC_WAIT(N)
#endif

static __device__ __forceinline__ float sigf(float x) {
    return 1.0f / (1.0f + __expf(-x));
}

// A-fragment (16x32 f16): lane m = lane&15; kb = (lane<16?0:8) relative to the
// 32-wide K window.  Halves 0..15 = K 2*kb..2*kb+15 (contiguous, matches B).
static __device__ __forceinline__ v16h load_a_frag(const _Float16* __restrict__ row,
                                                   int kb) {
    v8h lo = *(const v8h*)(row + 2 * kb);
    v8h hi = *(const v8h*)(row + 2 * kb + 8);
    v16h a;
#pragma unroll
    for (int i = 0; i < 8; ++i) { a[i] = lo[i]; a[8 + i] = hi[i]; }
    return a;
}

// ---------------------------------------------------------------------------
// fp32 -> fp16 conversion (grid-stride)
// ---------------------------------------------------------------------------
__global__ void k_f32_to_f16(const float* __restrict__ src,
                             _Float16* __restrict__ dst, size_t n) {
    size_t i = (size_t)blockIdx.x * blockDim.x + threadIdx.x;
    size_t stride = (size_t)gridDim.x * blockDim.x;
    for (; i < n; i += stride) dst[i] = (_Float16)src[i];
}

__global__ void k_bias_sum(const float* __restrict__ a,
                           const float* __restrict__ b,
                           float* __restrict__ dst, int n) {
    int i = blockIdx.x * blockDim.x + threadIdx.x;
    if (i < n) dst[i] = a[i] + b[i];
}

// ---------------------------------------------------------------------------
// Input GEMM: pre[btile, s, nt] (fragment order) = X . Wih^T + bsum
//   X   : fp16, row (b*S + s), D columns
//   Wih : fp16, [4H rows, D cols] row-major  (B columns of the GEMM)
//   pre : fp16 fragments, index ((btile*S + s)*NT + nt)*256 + lane*8 + r
// One wave per 16x16 output tile; KT = D/32 WMMA steps.
// ---------------------------------------------------------------------------
template <int KT>
__global__ void k_input_gemm(const _Float16* __restrict__ X,
                             const _Float16* __restrict__ Wih,
                             const float* __restrict__ bsum,
                             _Float16* __restrict__ pre,
                             int NT) {
    const int lane = threadIdx.x & 31;
    const int wid  = threadIdx.x >> 5;
    const int id   = blockIdx.x * 8 + wid;           // global tile id
    const int nt   = id % NT;
    const int s    = (id / NT) % SS;
    const int bt   = id / (NT * SS);

    const int n    = lane & 15;
    const int half = lane >> 4;
    const int kb   = half * 8;
    const int ks   = half * 16;
    const int D    = KT * 32;

    const _Float16* arow = X + ((size_t)(bt * 16 + n) * SS + s) * D;  // A row m = lane&15
    const _Float16* brow = Wih + (size_t)(nt * 16 + n) * D;           // B col n = Wih row

    v8f acc = {};
#pragma unroll
    for (int kt = 0; kt < KT; ++kt) {
        v16h a = load_a_frag(arow + kt * 32, kb);
        v16h b = *(const v16h*)(brow + kt * 32 + ks);
        acc = __builtin_amdgcn_wmma_f32_16x16x32_f16(
            false, a, false, b, (short)0, acc, false, false);
    }

    const float bs = bsum[nt * 16 + n];
    v8h out;
#pragma unroll
    for (int r = 0; r < 8; ++r) out[r] = (_Float16)(acc[r] + bs);

    *(v8h*)(pre + (((size_t)bt * SS + s) * NT + (size_t)nt) * 256 + lane * 8) = out;
}

// ---------------------------------------------------------------------------
// Persistent LSTM recurrence.  grid = 16 (batch tiles), block = (H/32)*32.
// Wave w owns hidden slice [w*32, w*32+32): 8 gate tiles (i,f,g,o x 2).
// Dynamic LDS layout (halves):
//   [0,                2*16*H)              h double buffer
//   [2*16*H,           2*16*H + 2*NT*256)   pre-fragment staging (2 buffers)
// Each wave async-prefetches its own 8 fragment tiles for step s+1 while
// computing step s; s_wait_asynccnt(8) retires the older batch.
// ---------------------------------------------------------------------------
template <int H>
__global__ void k_lstm_recurrent(const _Float16* __restrict__ Whh,  // [4H][H] fp16
                                 const _Float16* __restrict__ pre,  // fragments
                                 _Float16* __restrict__ hseq) {     // [(b*S+s)*H]
    constexpr int KT   = H / 32;      // K tiles per gate row
    constexpr int NT   = 4 * H / 16;  // gate tiles total
    constexpr int NH16 = H / 16;

    extern __shared__ _Float16 smem[];
    _Float16* hbuf   = smem;               // [2][16*H]
    _Float16* pstage = smem + 2 * 16 * H;  // [2][NT*256]

    const int lane = threadIdx.x & 31;
    const int w    = threadIdx.x >> 5;
    const int bt   = blockIdx.x;
    const int b0   = bt * 16;
    const int j0   = w * 32;

    const int n    = lane & 15;
    const int half = lane >> 4;
    const int kb   = half * 8;
    const int ks   = half * 16;
    const int mofs = half * 8;  // C-frag: M = r + 8*(lane>=16)

    // zero both h buffers (h_0 = 0)
    for (int i = threadIdx.x; i < 2 * 16 * H; i += blockDim.x)
        hbuf[i] = (_Float16)0;
    __syncthreads();

    // Tile n-tile indices (within the NT gate tiles) owned by this wave.
    int ntidx[8];
    const _Float16* bptr[8];
#pragma unroll
    for (int t = 0; t < 8; ++t) {
        const int gi = t >> 1, st = t & 1;
        ntidx[t] = gi * NH16 + (j0 >> 4) + st;
        bptr[t]  = Whh + (size_t)((gi * H) + j0 + st * 16 + n) * H;
    }

    const _Float16* pbase = pre + (size_t)bt * SS * NT * 256;

    // ---- prologue: async-prefetch step 0 fragments into pstage buffer 0 ----
#pragma unroll
    for (int t = 0; t < 8; ++t) {
        const size_t o = (size_t)ntidx[t] * 256 + lane * 8;
        async_copy16(pbase + o, pstage + o);
    }

    float c[2][8];
#pragma unroll
    for (int st = 0; st < 2; ++st)
#pragma unroll
        for (int r = 0; r < 8; ++r) c[st][r] = 0.0f;

    for (int s = 0; s < SS; ++s) {
        const int p = s & 1;

        // ---- async-prefetch step s+1, then retire the batch for step s ----
        if (s + 1 < SS) {
#pragma unroll
            for (int t = 0; t < 8; ++t) {
                const size_t o = (size_t)ntidx[t] * 256 + lane * 8;
                async_copy16(pbase + (size_t)(s + 1) * NT * 256 + o,
                             pstage + (size_t)(p ^ 1) * NT * 256 + o);
            }
            ASYNC_WAIT(8);
        } else {
            ASYNC_WAIT(0);
        }

        // ---- init accumulators from staged input-GEMM fragments ----
        const _Float16* pf = pstage + (size_t)p * NT * 256;
        v8f acc[8];
#pragma unroll
        for (int t = 0; t < 8; ++t) {
            v8h pv = *(const v8h*)(pf + (size_t)ntidx[t] * 256 + lane * 8);
#pragma unroll
            for (int r = 0; r < 8; ++r) acc[t][r] = (float)pv[r];
        }

        // ---- recurrent GEMM:  acc += h_{s-1} . Whh^T ----
#pragma unroll
        for (int kt = 0; kt < KT; ++kt) {
            const _Float16* arow = &hbuf[(size_t)p * 16 * H + (lane & 15) * H + kt * 32];
            v16h a = load_a_frag(arow, kb);
#pragma unroll
            for (int t = 0; t < 8; ++t) {
                v16h b = *(const v16h*)(bptr[t] + kt * 32 + ks);
                acc[t] = __builtin_amdgcn_wmma_f32_16x16x32_f16(
                    false, a, false, b, (short)0, acc[t], false, false);
            }
        }

        // ---- gate activations + state update (lane-local) ----
        _Float16* hw = &hbuf[(size_t)(p ^ 1) * 16 * H];
#pragma unroll
        for (int st = 0; st < 2; ++st) {
#pragma unroll
            for (int r = 0; r < 8; ++r) {
                const float iv = sigf(acc[0 + st][r]);
                const float fv = sigf(acc[2 + st][r]);
                const float gv = tanhf(acc[4 + st][r]);
                const float ov = sigf(acc[6 + st][r]);
                const float cn = fv * c[st][r] + iv * gv;
                c[st][r] = cn;
                const float hv = ov * tanhf(cn);
                const int m = r + mofs;
                const int j = j0 + st * 16 + n;
                const _Float16 h16 = (_Float16)hv;
                hw[m * H + j] = h16;
                hseq[((size_t)(b0 + m) * SS + s) * H + j] = h16;
            }
        }
        __syncthreads();  // publish h_s before any wave reads it at step s+1
    }
}

// ---------------------------------------------------------------------------
// Head: out[b,s] = tanh(dot(h2[b,s,:], Wl) + bl)
// ---------------------------------------------------------------------------
__global__ void k_head(const _Float16* __restrict__ h2,
                       const float* __restrict__ Wl,
                       const float* __restrict__ bl,
                       float* __restrict__ out, size_t rows) {
    size_t t = (size_t)blockIdx.x * blockDim.x + threadIdx.x;
    if (t >= rows) return;
    const _Float16* r = h2 + t * 256;
    float acc = 0.0f;
#pragma unroll 4
    for (int j = 0; j < 256; j += 8) {
        v8h v = *(const v8h*)(r + j);
#pragma unroll
        for (int k = 0; k < 8; ++k) acc += (float)v[k] * Wl[j + k];
    }
    out[t] = tanhf(acc + bl[0]);
}

// ---------------------------------------------------------------------------
// Launch
// ---------------------------------------------------------------------------
extern "C" void kernel_launch(void* const* d_in, const int* in_sizes, int n_in,
                              void* d_out, int out_size, void* d_ws, size_t ws_size,
                              hipStream_t stream) {
    (void)in_sizes; (void)n_in; (void)out_size; (void)ws_size;

    const float* noise = (const float*)d_in[0];
    const float* Wih[3] = {(const float*)d_in[1], (const float*)d_in[5], (const float*)d_in[9]};
    const float* Whh[3] = {(const float*)d_in[2], (const float*)d_in[6], (const float*)d_in[10]};
    const float* bih[3] = {(const float*)d_in[3], (const float*)d_in[7], (const float*)d_in[11]};
    const float* bhh[3] = {(const float*)d_in[4], (const float*)d_in[8], (const float*)d_in[12]};
    const float* Wl = (const float*)d_in[13];
    const float* bl = (const float*)d_in[14];
    float* out = (float*)d_out;

    // ---- carve workspace ----
    size_t off = 0;
    auto carve = [&](size_t bytes) -> char* {
        char* p = (char*)d_ws + off;
        off += (bytes + 255) & ~(size_t)255;
        return p;
    };
    const size_t ROWS = (size_t)BB * SS;
    _Float16* x16 = (_Float16*)carve(ROWS * 32 * 2);
    _Float16* hs[3];
    hs[0] = (_Float16*)carve(ROWS * 64 * 2);
    hs[1] = (_Float16*)carve(ROWS * 128 * 2);
    hs[2] = (_Float16*)carve(ROWS * 256 * 2);
    const int Hdim[3] = {64, 128, 256};
    const int Din[3]  = {32, 64, 128};
    _Float16 *wih16[3], *whh16[3];
    float* bsum[3];
    for (int l = 0; l < 3; ++l) {
        wih16[l] = (_Float16*)carve((size_t)4 * Hdim[l] * Din[l] * 2);
        whh16[l] = (_Float16*)carve((size_t)4 * Hdim[l] * Hdim[l] * 2);
        bsum[l]  = (float*)carve((size_t)4 * Hdim[l] * 4);
    }
    _Float16* pre = (_Float16*)carve(ROWS * 1024 * 2);  // reused, sized for layer 2

    // ---- fp16 conversions ----
    k_f32_to_f16<<<4096, 256, 0, stream>>>(noise, x16, ROWS * 32);
    for (int l = 0; l < 3; ++l) {
        const size_t nih = (size_t)4 * Hdim[l] * Din[l];
        const size_t nhh = (size_t)4 * Hdim[l] * Hdim[l];
        k_f32_to_f16<<<(unsigned)((nih + 255) / 256), 256, 0, stream>>>(Wih[l], wih16[l], nih);
        k_f32_to_f16<<<(unsigned)((nhh + 255) / 256), 256, 0, stream>>>(Whh[l], whh16[l], nhh);
        k_bias_sum<<<(4 * Hdim[l] + 255) / 256, 256, 0, stream>>>(bih[l], bhh[l], bsum[l], 4 * Hdim[l]);
    }

    auto smem_bytes = [](int H) -> size_t {
        const int NT = 4 * H / 16;
        return (size_t)(2 * 16 * H + 2 * NT * 256) * sizeof(_Float16);
    };

    // ---- layer 0: D=32 (KT=1), H=64 ----
    {
        const int NT = 4 * 64 / 16;  // 16
        k_input_gemm<1><<<16 * SS * NT / 8, 256, 0, stream>>>(x16, wih16[0], bsum[0], pre, NT);
        k_lstm_recurrent<64><<<16, 64, smem_bytes(64), stream>>>(whh16[0], pre, hs[0]);
    }
    // ---- layer 1: D=64 (KT=2), H=128 ----
    {
        const int NT = 4 * 128 / 16;  // 32
        k_input_gemm<2><<<16 * SS * NT / 8, 256, 0, stream>>>(hs[0], wih16[1], bsum[1], pre, NT);
        k_lstm_recurrent<128><<<16, 128, smem_bytes(128), stream>>>(whh16[1], pre, hs[1]);
    }
    // ---- layer 2: D=128 (KT=4), H=256 ----
    {
        const int NT = 4 * 256 / 16;  // 64
        k_input_gemm<4><<<16 * SS * NT / 8, 256, 0, stream>>>(hs[1], wih16[2], bsum[2], pre, NT);
        k_lstm_recurrent<256><<<16, 256, smem_bytes(256), stream>>>(whh16[2], pre, hs[2]);
    }

    // ---- head ----
    k_head<<<(unsigned)((ROWS + 255) / 256), 256, 0, stream>>>(hs[2], Wl, bl, out, ROWS);
}
